// GPT2Attention2Llama_15960098472558
// MI455X (gfx1250) — compile-verified, matrix operand order checked by reading, not run on hardware
//
#include <hip/hip_runtime.h>

// ---------------- problem constants ----------------
constexpr int Bb = 4;
constexpr int S  = 2048;
constexpr int D  = 1024;
constexpr int H  = 16;
constexpr int HD = 64;
constexpr int M  = Bb * S;   // 8192 rows

typedef __attribute__((ext_vector_type(16))) __bf16 v16bf;
typedef __attribute__((ext_vector_type(8)))  float  v8f;

struct Frag32B { unsigned int d[8]; };   // 16 bf16 = 32 bytes

// round-to-nearest-even f32 -> bf16
__device__ __forceinline__ unsigned short f2bf(float f) {
    unsigned int u = __builtin_bit_cast(unsigned int, f);
    unsigned int r = u + 0x7FFFu + ((u >> 16) & 1u);
    return (unsigned short)(r >> 16);
}

// Load one 16-element bf16 fragment row: elements 0..7 from p0, 8..15 from p1.
// Matches ISA A/B layout: per-lane k-chunks [koff,koff+8) and [koff+16,koff+24).
__device__ __forceinline__ v16bf load_frag(const unsigned short* p0,
                                           const unsigned short* p1) {
    uint4 a = *(const uint4*)p0;
    uint4 b = *(const uint4*)p1;
    Frag32B f = {{a.x, a.y, a.z, a.w, b.x, b.y, b.z, b.w}};
    return __builtin_bit_cast(v16bf, f);
}

__device__ __forceinline__ v8f wmma_bf16(v16bf a, v16bf b, v8f c) {
    return __builtin_amdgcn_wmma_f32_16x16x32_bf16(
        /*neg_a=*/false, a, /*neg_b=*/false, b,
        /*c_mod=*/(short)0, c, /*reuse_a=*/false, /*reuse_b=*/false);
}

// ---------------- f32 -> bf16 conversion ----------------
__global__ void f32_to_bf16_kernel(const float* __restrict__ src,
                                   unsigned short* __restrict__ dst, int n4) {
    int i = blockIdx.x * blockDim.x + threadIdx.x;
    if (i < n4) {
        float4 f = ((const float4*)src)[i];
        unsigned int lo = (unsigned int)f2bf(f.x) | ((unsigned int)f2bf(f.y) << 16);
        unsigned int hi = (unsigned int)f2bf(f.z) | ((unsigned int)f2bf(f.w) << 16);
        ((uint2*)dst)[i] = make_uint2(lo, hi);
    }
}

// ---------------- GEMM fragment bundle (32x64 wave tile, one 32-K step) ------
struct GemmFrags {
    v16bf a0, a1;    // two 16x32 A fragments (rows m0..m0+31)
    v16bf b[4];      // four 32x16 B fragments (cols n0..n0+63)
};

__device__ __forceinline__ GemmFrags gemm_load_step(
    const unsigned short* __restrict__ xb, const unsigned short* __restrict__ wb,
    int m0, int n0, int kb, int lr, int koff) {
    GemmFrags f;
    const unsigned short* ar0 = xb + (size_t)(m0 + lr) * D + kb + koff;
    const unsigned short* ar1 = xb + (size_t)(m0 + 16 + lr) * D + kb + koff;
    f.a0 = load_frag(ar0, ar0 + 16);
    f.a1 = load_frag(ar1, ar1 + 16);
#pragma unroll
    for (int ni = 0; ni < 4; ni++) {
        const unsigned short* wr = wb + (size_t)(n0 + ni * 16 + lr) * D + kb + koff;
        f.b[ni] = load_frag(wr, wr + 16);
    }
    return f;
}

// Software-pipelined 32x64 GEMM accumulate over K=D: next K-step's 12 b128
// loads are in flight while the current step's 8 WMMAs execute, so the
// pre-WMMA wait is partial instead of s_wait_loadcnt 0.
__device__ __forceinline__ void gemm_acc(
    const unsigned short* __restrict__ xb, const unsigned short* __restrict__ wb,
    int m0, int n0, int lr, int koff, v8f acc[2][4]) {
    GemmFrags cur = gemm_load_step(xb, wb, m0, n0, 0, lr, koff);
    for (int kb = 0; kb < D; kb += 32) {
        const int kn = (kb + 32 < D) ? kb + 32 : kb;   // clamped: branch-free tail
        GemmFrags nxt = gemm_load_step(xb, wb, m0, n0, kn, lr, koff);
#pragma unroll
        for (int ni = 0; ni < 4; ni++) {
            acc[0][ni] = wmma_bf16(cur.a0, cur.b[ni], acc[0][ni]);
            acc[1][ni] = wmma_bf16(cur.a1, cur.b[ni], acc[1][ni]);
        }
        cur = nxt;
    }
}

// ---------------- fused QKV projection: y = x @ W^T + b ----------------
// block = 128 threads (4 waves); wave tile = 32(M) x 64(N); grid.z selects Q/K/V.
// Q scaled by 1/sqrt(HD) and stored [B,H,S,HD]; K stored [B,H,S,HD]; V stored
// transposed [B,H,HD,S] so the P@V stage reads K-contiguous B fragments.
__global__ __launch_bounds__(128) void qkv_kernel(
    const unsigned short* __restrict__ xb,
    const unsigned short* __restrict__ wqb, const unsigned short* __restrict__ wkb,
    const unsigned short* __restrict__ wvb,
    const float* __restrict__ bqf, const float* __restrict__ bkf,
    const float* __restrict__ bvf,
    unsigned short* __restrict__ qo, unsigned short* __restrict__ ko,
    unsigned short* __restrict__ vo) {
    const int z    = blockIdx.z;
    const unsigned short* wb = (z == 0) ? wqb : (z == 1) ? wkb : wvb;
    const float* bias        = (z == 0) ? bqf : (z == 1) ? bkf : bvf;

    const int lane = threadIdx.x & 31;
    const int wave = threadIdx.x >> 5;
    const int m0   = blockIdx.x * 128 + wave * 32;
    const int n0   = blockIdx.y * 64;
    const int lr   = lane & 15;
    const int hi   = lane >> 4;
    const int koff = hi * 8;

    v8f acc[2][4] = {};
    gemm_acc(xb, wb, m0, n0, lr, koff, acc);

    const float qscale = (z == 0) ? 0.125f : 1.0f;  // 1/sqrt(64), folded incl. bias
    unsigned short* qk_out = (z == 0) ? qo : ko;
#pragma unroll
    for (int mi = 0; mi < 2; mi++)
#pragma unroll
        for (int ni = 0; ni < 4; ni++)
#pragma unroll
            for (int r = 0; r < 8; r++) {
                int m = m0 + mi * 16 + r + hi * 8;
                int c = n0 + ni * 16 + lr;
                float v = (acc[mi][ni][r] + bias[c]) * qscale;
                int bidx = m >> 11;        // / S
                int s    = m & (S - 1);
                int h    = c >> 6;         // / HD
                int dd   = c & (HD - 1);
                unsigned short bv = f2bf(v);
                if (z == 2)
                    vo[(size_t)((bidx * H + h) * HD + dd) * S + s] = bv;   // V^T
                else
                    qk_out[(size_t)((bidx * H + h) * S + s) * HD + dd] = bv;
            }
}

// ---------------- flash attention ----------------
// block = 128 threads (4 independent waves); wave handles one 16-query tile of
// one (b,h). Key loop in steps of 32 with causal early-exit; online softmax;
// P goes through LDS for C-layout -> A-layout transpose.
__global__ __launch_bounds__(128) void attn_kernel(
    const unsigned short* __restrict__ qg, const unsigned short* __restrict__ kg,
    const unsigned short* __restrict__ vtg, unsigned short* __restrict__ aog) {
    __shared__ __align__(16) unsigned short pshm[4 * 16 * 32];

    const int lane = threadIdx.x & 31;
    const int wave = threadIdx.x >> 5;
    const int bh   = blockIdx.y;                      // b*H + h
    const int q0   = (blockIdx.x * 4 + wave) * 16;
    const int lr   = lane & 15;
    const int hi   = lane >> 4;
    const int koff = hi * 8;

    const unsigned short* Q  = qg  + (size_t)bh * S * HD;
    const unsigned short* K  = kg  + (size_t)bh * S * HD;
    const unsigned short* Vt = vtg + (size_t)bh * HD * S;
    unsigned short* pw = pshm + wave * 16 * 32;

    const unsigned short* qr = Q + (size_t)(q0 + lr) * HD;
    v16bf aq0 = load_frag(qr + koff,      qr + koff + 16);
    v16bf aq1 = load_frag(qr + 32 + koff, qr + 32 + koff + 16);

    float mrow[8], lrow[8];
#pragma unroll
    for (int r = 0; r < 8; r++) { mrow[r] = -3.0e38f; lrow[r] = 0.0f; }
    v8f oacc[4] = {};

    for (int kbv = 0; kbv <= q0 + 15; kbv += 32) {
        // prefetch next key block's K rows into cache (global_prefetch_b8)
        __builtin_prefetch(K + (size_t)(kbv + 32 + lr) * HD, 0, 1);
        __builtin_prefetch(K + (size_t)(kbv + 48 + lr) * HD, 0, 1);

        // scores: two 16x16 tiles over this 32-key block
        v8f s0 = {}, s1 = {};
        {
            const unsigned short* kr0 = K + (size_t)(kbv + lr) * HD;
            v16bf b00 = load_frag(kr0 + koff,      kr0 + koff + 16);
            v16bf b01 = load_frag(kr0 + 32 + koff, kr0 + 32 + koff + 16);
            s0 = wmma_bf16(aq0, b00, s0);
            s0 = wmma_bf16(aq1, b01, s0);
            const unsigned short* kr1 = K + (size_t)(kbv + 16 + lr) * HD;
            v16bf b10 = load_frag(kr1 + koff,      kr1 + koff + 16);
            v16bf b11 = load_frag(kr1 + 32 + koff, kr1 + 32 + koff + 16);
            s1 = wmma_bf16(aq0, b10, s1);
            s1 = wmma_bf16(aq1, b11, s1);
        }
        // V fragments are independent of the softmax chain -> loads overlap it
        v16bf bv[4];
#pragma unroll
        for (int ni = 0; ni < 4; ni++) {
            const unsigned short* vr = Vt + (size_t)(ni * 16 + lr) * S + kbv;
            bv[ni] = load_frag(vr + koff, vr + koff + 16);
        }
        // causal mask + online softmax (row-wise across the 16-lane C groups)
#pragma unroll
        for (int r = 0; r < 8; r++) {
            int qrow = q0 + r + hi * 8;
            float e0 = ((kbv + lr)      <= qrow) ? s0[r] : -10000.0f;
            float e1 = ((kbv + 16 + lr) <= qrow) ? s1[r] : -10000.0f;
            float t = fmaxf(e0, e1);
#pragma unroll
            for (int off = 1; off < 16; off <<= 1)
                t = fmaxf(t, __shfl_xor(t, off, 32));
            float mnew  = fmaxf(mrow[r], t);
            float alpha = __expf(mrow[r] - mnew);
            float p0 = __expf(e0 - mnew);
            float p1 = __expf(e1 - mnew);
            float ps = p0 + p1;
#pragma unroll
            for (int off = 1; off < 16; off <<= 1)
                ps += __shfl_xor(ps, off, 32);
            lrow[r] = lrow[r] * alpha + ps;
            mrow[r] = mnew;
#pragma unroll
            for (int ni = 0; ni < 4; ni++) oacc[ni][r] *= alpha;
            int prow = r + hi * 8;
            pw[prow * 32 + lr]      = f2bf(p0);
            pw[prow * 32 + 16 + lr] = f2bf(p1);
        }
        // intra-wave LDS is in-order; fence against compiler reordering + drain DS
        asm volatile("s_wait_dscnt 0" ::: "memory");
        // P (16x32) as A fragment from LDS
        v16bf ap = load_frag(pw + lr * 32 + koff, pw + lr * 32 + koff + 16);
        // P @ V using transposed V: B[k][n] = Vt[dim][kbv+k], K-contiguous
#pragma unroll
        for (int ni = 0; ni < 4; ni++)
            oacc[ni] = wmma_bf16(ap, bv[ni], oacc[ni]);
    }

    // normalize + store attn output as bf16 in [B,S,D]
    const int bidx = bh >> 4;     // / H
    const int h    = bh & 15;
#pragma unroll
    for (int ni = 0; ni < 4; ni++)
#pragma unroll
        for (int r = 0; r < 8; r++) {
            int srow = q0 + r + hi * 8;
            int col  = h * HD + ni * 16 + lr;
            float v  = oacc[ni][r] / lrow[r];
            aog[(size_t)(bidx * S + srow) * D + col] = f2bf(v);
        }
}

// ---------------- output projection: out = A @ Wo^T + bo + comp ----------------
__global__ __launch_bounds__(128) void outproj_kernel(
    const unsigned short* __restrict__ ab, const unsigned short* __restrict__ wob,
    const float* __restrict__ bo, const float* __restrict__ comp,
    float* __restrict__ out) {
    const int lane = threadIdx.x & 31;
    const int wave = threadIdx.x >> 5;
    const int m0   = blockIdx.x * 128 + wave * 32;
    const int n0   = blockIdx.y * 64;
    const int lr   = lane & 15;
    const int hi   = lane >> 4;
    const int koff = hi * 8;

    v8f acc[2][4] = {};
    gemm_acc(ab, wob, m0, n0, lr, koff, acc);

#pragma unroll
    for (int mi = 0; mi < 2; mi++)
#pragma unroll
        for (int ni = 0; ni < 4; ni++)
#pragma unroll
            for (int r = 0; r < 8; r++) {
                int m = m0 + mi * 16 + r + hi * 8;
                int c = n0 + ni * 16 + lr;
                out[(size_t)m * D + c] = acc[mi][ni][r] + bo[c] + comp[c];
            }
}

// ---------------- host launch ----------------
extern "C" void kernel_launch(void* const* d_in, const int* in_sizes, int n_in,
                              void* d_out, int out_size, void* d_ws, size_t ws_size,
                              hipStream_t stream) {
    const float* x    = (const float*)d_in[0];
    const float* Wq   = (const float*)d_in[1];
    const float* bq   = (const float*)d_in[2];
    const float* Wk   = (const float*)d_in[3];
    const float* bk   = (const float*)d_in[4];
    const float* Wv   = (const float*)d_in[5];
    const float* bv   = (const float*)d_in[6];
    const float* Wo   = (const float*)d_in[7];
    const float* bo   = (const float*)d_in[8];
    const float* comp = (const float*)d_in[9];
    float* out = (float*)d_out;

    char* ws = (char*)d_ws;
    unsigned short* xb  = (unsigned short*)ws; ws += (size_t)M * D * 2;   // 16 MB
    unsigned short* wqb = (unsigned short*)ws; ws += (size_t)D * D * 2;   //  2 MB
    unsigned short* wkb = (unsigned short*)ws; ws += (size_t)D * D * 2;
    unsigned short* wvb = (unsigned short*)ws; ws += (size_t)D * D * 2;
    unsigned short* wob = (unsigned short*)ws; ws += (size_t)D * D * 2;
    unsigned short* qb  = (unsigned short*)ws; ws += (size_t)M * D * 2;   // [B,H,S,HD]
    unsigned short* kb  = (unsigned short*)ws; ws += (size_t)M * D * 2;   // [B,H,S,HD]
    unsigned short* vtb = (unsigned short*)ws; ws += (size_t)M * D * 2;   // [B,H,HD,S]
    unsigned short* aob = (unsigned short*)ws; ws += (size_t)M * D * 2;   // [B,S,D]

    const int xn4 = M * D / 4, wn4 = D * D / 4;
    f32_to_bf16_kernel<<<(xn4 + 255) / 256, 256, 0, stream>>>(x,  xb,  xn4);
    f32_to_bf16_kernel<<<(wn4 + 255) / 256, 256, 0, stream>>>(Wq, wqb, wn4);
    f32_to_bf16_kernel<<<(wn4 + 255) / 256, 256, 0, stream>>>(Wk, wkb, wn4);
    f32_to_bf16_kernel<<<(wn4 + 255) / 256, 256, 0, stream>>>(Wv, wvb, wn4);
    f32_to_bf16_kernel<<<(wn4 + 255) / 256, 256, 0, stream>>>(Wo, wob, wn4);

    dim3 gq(M / 128, D / 64, 3);
    qkv_kernel<<<gq, 128, 0, stream>>>(xb, wqb, wkb, wvb, bq, bk, bv, qb, kb, vtb);

    dim3 ga(S / 64, Bb * H);
    attn_kernel<<<ga, 128, 0, stream>>>(qb, kb, vtb, aob);

    dim3 go(M / 128, D / 64);
    outproj_kernel<<<go, 128, 0, stream>>>(aob, wob, bo, comp, out);
}
